// AttnDecoderRNN_4595615007406
// MI455X (gfx1250) — matrix operand order repaired and checked by reading, compiled, not run on hardware
//
#include <hip/hip_runtime.h>
#include <hip/hip_bf16.h>
#include <math.h>

#define H_DIM 1024
#define B_DIM 512
#define L_DIM 15
#define V_DIM 32000

typedef __attribute__((ext_vector_type(16))) __bf16 v16bf;
typedef __attribute__((ext_vector_type(8)))  float  v8f;

union AFrag { v16bf v; uint4 q[2]; };

// ---------------- block reduction (wave32 shuffle + LDS) ----------------
__device__ __forceinline__ float block_reduce(float v, bool isMax) {
  __shared__ float sp[8];
  int lane = threadIdx.x & 31, w = threadIdx.x >> 5;
  #pragma unroll
  for (int off = 16; off; off >>= 1) {
    float o = __shfl_xor(v, off, 32);
    v = isMax ? fmaxf(v, o) : v + o;
  }
  if (lane == 0) sp[w] = v;
  __syncthreads();
  int nw = blockDim.x >> 5;
  if (w == 0) {
    float t = (lane < nw) ? sp[lane] : (isMax ? -INFINITY : 0.0f);
    #pragma unroll
    for (int off = 16; off; off >>= 1) {
      float o = __shfl_xor(t, off, 32);
      t = isMax ? fmaxf(t, o) : t + o;
    }
    if (lane == 0) sp[0] = t;
  }
  __syncthreads();
  float r = sp[0];
  __syncthreads();
  return r;
}

// ---------------- K1: gather + additive attention + context ----------------
__global__ __launch_bounds__(256)
void attn_kernel(const int* __restrict__ ids,
                 const float* __restrict__ hidden,
                 const float* __restrict__ enc,
                 const float* __restrict__ embedding,
                 const float* __restrict__ attn1_w,
                 const float* __restrict__ attn1_b,
                 __bf16* __restrict__ cat_bf16,   // [B][2H]
                 __bf16* __restrict__ h_bf16,     // [B][H]
                 float* __restrict__ attn_out)    // [L][B]
{
  const int b = blockIdx.x, tid = threadIdx.x;
  __shared__ float s_emb[H_DIM];
  const int id = ids[b];
  for (int h = tid; h < H_DIM; h += 256) {
    float e  = embedding[(size_t)id * H_DIM + h];
    s_emb[h] = e;
    cat_bf16[(size_t)b * (2 * H_DIM) + h] = (__bf16)e;
    float hv = hidden[(size_t)b * H_DIM + h];
    h_bf16[(size_t)b * H_DIM + h] = (__bf16)hv;
  }
  __syncthreads();

  float acc[L_DIM + 1];
  #pragma unroll
  for (int i = 0; i <= L_DIM; ++i) acc[i] = 0.0f;

  for (int h = tid; h < H_DIM; h += 256) {
    float we = attn1_w[h], wc = attn1_w[H_DIM + h];
    acc[L_DIM] += s_emb[h] * we;
    #pragma unroll
    for (int l = 0; l < L_DIM; ++l)
      acc[l] += enc[(size_t)l * B_DIM * H_DIM + (size_t)b * H_DIM + h] * wc;
  }
  float red[L_DIM + 1];
  for (int i = 0; i <= L_DIM; ++i) red[i] = block_reduce(acc[i], false);

  const float bias = attn1_b[0] + red[L_DIM];
  float mx = -INFINITY;
  #pragma unroll
  for (int l = 0; l < L_DIM; ++l) mx = fmaxf(mx, red[l] + bias);
  float sum = 0.0f, aw[L_DIM];
  #pragma unroll
  for (int l = 0; l < L_DIM; ++l) { aw[l] = expf(red[l] + bias - mx); sum += aw[l]; }
  float inv = 1.0f / sum;
  #pragma unroll
  for (int l = 0; l < L_DIM; ++l) aw[l] *= inv;
  if (tid < L_DIM) attn_out[(size_t)tid * B_DIM + b] = aw[tid];

  for (int h = tid; h < H_DIM; h += 256) {
    float cv = 0.0f;
    #pragma unroll
    for (int l = 0; l < L_DIM; ++l)
      cv += aw[l] * enc[(size_t)l * B_DIM * H_DIM + (size_t)b * H_DIM + h];
    cat_bf16[(size_t)b * (2 * H_DIM) + H_DIM + h] = (__bf16)cv;
  }
}

// ---------------- WMMA GEMM: out[M][N] = A_bf16[M][K] * W_f32[N][K]^T + bias ----------------
// wave tile = 64(M) x 64(N): 4 A-frags x 4 B-frags = 16 v_wmma_f32_16x16x32_bf16 per K-chunk.
// Weight f32->bf16 conversion (one cvt per element) hides under XDL co-execution.
// EPI 0: f32 store (+bias). EPI 1: relu -> bf16 store (+bias).
template <int EPI>
__global__ __launch_bounds__(256)
void gemm_bf16_wmma(const __bf16* __restrict__ A,
                    const float* __restrict__ W,
                    const float* __restrict__ bias,
                    float* __restrict__ outF,
                    __bf16* __restrict__ outB,
                    int M, int N, int K)
{
  const int gw   = blockIdx.x * (blockDim.x >> 5) + (threadIdx.x >> 5);
  const int lane = threadIdx.x & 31;
  const int mtiles = M >> 6, ngroups = N >> 6;
  if (gw >= mtiles * ngroups) return;            // whole-wave exit: EXEC stays all-ones
  const int mi = gw % mtiles, ni = gw / mtiles;  // M-fastest: concurrent waves share W rows in L2
  const int m0 = mi << 6, n0 = ni << 6;
  const int r = lane & 15, half = lane >> 4;

  v8f c[16];
  #pragma unroll
  for (int i = 0; i < 16; ++i) c[i] = v8f{};

  for (int k0 = 0; k0 < K; k0 += 32) {
    // A fragments: 16x32 bf16 each; per lane two contiguous 16B runs
    AFrag a[4];
    #pragma unroll
    for (int mm = 0; mm < 4; ++mm) {
      const __bf16* ap = A + (size_t)(m0 + mm * 16 + r) * K + k0 + half * 8;
      a[mm].q[0] = *reinterpret_cast<const uint4*>(ap);
      a[mm].q[1] = *reinterpret_cast<const uint4*>(ap + 16);
    }
    #pragma unroll
    for (int j = 0; j < 4; ++j) {
      const int n = n0 + j * 16 + r;
      const float* wrow = W + (size_t)n * K + k0 + (half << 3);
      if (k0 + 32 < K) __builtin_prefetch(wrow + 32, 0, 1);  // global_prefetch_b8: next chunk
      // B fragment: element e <-> K = half*8 + e (e<8), 16 + half*8 + (e-8) (e>=8) (wrow already offset by half*8)
      const float4* wq = reinterpret_cast<const float4*>(wrow);
      float4 q0 = wq[0], q1 = wq[1], q2 = wq[4], q3 = wq[5];
      v16bf bv;
      bv[0]  = (__bf16)q0.x; bv[1]  = (__bf16)q0.y; bv[2]  = (__bf16)q0.z; bv[3]  = (__bf16)q0.w;
      bv[4]  = (__bf16)q1.x; bv[5]  = (__bf16)q1.y; bv[6]  = (__bf16)q1.z; bv[7]  = (__bf16)q1.w;
      bv[8]  = (__bf16)q2.x; bv[9]  = (__bf16)q2.y; bv[10] = (__bf16)q2.z; bv[11] = (__bf16)q2.w;
      bv[12] = (__bf16)q3.x; bv[13] = (__bf16)q3.y; bv[14] = (__bf16)q3.z; bv[15] = (__bf16)q3.w;
      #pragma unroll
      for (int mm = 0; mm < 4; ++mm) {
        c[mm * 4 + j] = __builtin_amdgcn_wmma_f32_16x16x32_bf16(
            false, a[mm].v, false, bv, (short)0, c[mm * 4 + j], false, false);
      }
    }
  }

  #pragma unroll
  for (int j = 0; j < 4; ++j) {
    const int n = n0 + j * 16 + r;
    const float bv = bias ? bias[n] : 0.0f;
    #pragma unroll
    for (int mm = 0; mm < 4; ++mm) {
      #pragma unroll
      for (int e = 0; e < 8; ++e) {
        const int m = m0 + mm * 16 + e + half * 8;  // f32 D layout: VGPR e -> M=e / e+8
        float val = c[mm * 4 + j][e] + bv;
        if (EPI == 1) {
          outB[(size_t)m * N + n] = (__bf16)fmaxf(val, 0.0f);
        } else {
          outF[(size_t)m * N + n] = val;
        }
      }
    }
  }
}

// ---------------- K3c: GRU gates (elementwise) ----------------
__global__ __launch_bounds__(256)
void gru_gates(const float* __restrict__ gi, const float* __restrict__ gh,
               const float* __restrict__ hidden,
               float* __restrict__ hnew_out, __bf16* __restrict__ hnew_bf16)
{
  int i = blockIdx.x * blockDim.x + threadIdx.x;
  if (i >= B_DIM * H_DIM) return;
  int b = i >> 10, h = i & (H_DIM - 1);
  size_t base = (size_t)b * (3 * H_DIM);
  float ir = gi[base + h], iz = gi[base + H_DIM + h], in_ = gi[base + 2 * H_DIM + h];
  float hr = gh[base + h], hz = gh[base + H_DIM + h], hn = gh[base + 2 * H_DIM + h];
  float rg = 1.0f / (1.0f + expf(-(ir + hr)));
  float zg = 1.0f / (1.0f + expf(-(iz + hz)));
  float ng = tanhf(in_ + rg * hn);
  float hv = hidden[i];
  float hv2 = (1.0f - zg) * ng + zg * hv;
  hnew_out[i] = hv2;
  hnew_bf16[i] = (__bf16)hv2;
}

// ---------------- K5: in-place log-softmax over V ----------------
__global__ __launch_bounds__(256)
void logsoftmax_kernel(float* __restrict__ out)
{
  const int b = blockIdx.x;
  float* row = out + (size_t)b * V_DIM;
  float m = -INFINITY;
  for (int v = threadIdx.x; v < V_DIM; v += 256) m = fmaxf(m, row[v]);
  m = block_reduce(m, true);
  float s = 0.0f;
  for (int v = threadIdx.x; v < V_DIM; v += 256) s += expf(row[v] - m);
  s = block_reduce(s, false);
  const float lse = m + logf(s);
  for (int v = threadIdx.x; v < V_DIM; v += 256) row[v] -= lse;
}

extern "C" void kernel_launch(void* const* d_in, const int* in_sizes, int n_in,
                              void* d_out, int out_size, void* d_ws, size_t ws_size,
                              hipStream_t stream) {
  (void)in_sizes; (void)n_in; (void)out_size; (void)ws_size;
  const int*   ids       = (const int*)d_in[0];
  const float* hidden    = (const float*)d_in[1];
  const float* enc       = (const float*)d_in[2];
  const float* embedding = (const float*)d_in[4];
  const float* attn1_w   = (const float*)d_in[5];
  const float* attn1_b   = (const float*)d_in[6];
  const float* combine_w = (const float*)d_in[7];
  const float* combine_b = (const float*)d_in[8];
  const float* w_ih      = (const float*)d_in[9];
  const float* w_hh      = (const float*)d_in[10];
  const float* b_ih      = (const float*)d_in[11];
  const float* b_hh      = (const float*)d_in[12];
  const float* out_w     = (const float*)d_in[13];
  const float* out_b     = (const float*)d_in[14];

  float* out_logits = (float*)d_out;                               // [B][V]
  float* out_hnew   = out_logits + (size_t)B_DIM * V_DIM;          // [1][B][H]
  float* out_attn   = out_hnew + (size_t)B_DIM * H_DIM;            // [L][B][1]

  char* ws = (char*)d_ws;
  size_t off = 0;
  auto carve = [&](size_t bytes) {
    char* p = ws + off;
    off = (off + bytes + 255) & ~(size_t)255;
    return p;
  };
  __bf16* cat_bf16  = (__bf16*)carve((size_t)B_DIM * 2 * H_DIM * sizeof(__bf16)); // 2 MB
  __bf16* x_bf16    = (__bf16*)carve((size_t)B_DIM * H_DIM * sizeof(__bf16));     // 1 MB
  __bf16* h_bf16    = (__bf16*)carve((size_t)B_DIM * H_DIM * sizeof(__bf16));     // 1 MB
  __bf16* hnew_bf16 = (__bf16*)carve((size_t)B_DIM * H_DIM * sizeof(__bf16));     // 1 MB
  float*  gi        = (float*)carve((size_t)B_DIM * 3 * H_DIM * sizeof(float));   // 6 MB
  float*  gh        = (float*)carve((size_t)B_DIM * 3 * H_DIM * sizeof(float));   // 6 MB

  attn_kernel<<<B_DIM, 256, 0, stream>>>(ids, hidden, enc, embedding,
                                         attn1_w, attn1_b, cat_bf16, h_bf16, out_attn);

  // x = relu([emb,ctx] @ combine_w^T + b) -> bf16   (M=512,N=1024,K=2048): 8x16 tiles
  gemm_bf16_wmma<1><<<(8 * 16) / 8, 256, 0, stream>>>(
      cat_bf16, combine_w, combine_b, nullptr, x_bf16, 512, 1024, 2048);

  // gi = x @ w_ih^T + b_ih   (M=512,N=3072,K=1024): 8x48 tiles
  gemm_bf16_wmma<0><<<(8 * 48) / 8, 256, 0, stream>>>(
      x_bf16, w_ih, b_ih, gi, nullptr, 512, 3072, 1024);

  // gh = h @ w_hh^T + b_hh
  gemm_bf16_wmma<0><<<(8 * 48) / 8, 256, 0, stream>>>(
      h_bf16, w_hh, b_hh, gh, nullptr, 512, 3072, 1024);

  gru_gates<<<(B_DIM * H_DIM + 255) / 256, 256, 0, stream>>>(gi, gh, hidden,
                                                             out_hnew, hnew_bf16);

  // logits = h_new @ out_w^T + out_b   (M=512,N=32000,K=1024): 8x500 tiles, 33.6 GFLOP
  gemm_bf16_wmma<0><<<(8 * 500) / 8, 256, 0, stream>>>(
      hnew_bf16, out_w, out_b, out_logits, nullptr, 512, 32000, 1024);

  logsoftmax_kernel<<<B_DIM, 256, 0, stream>>>(out_logits);
}
